// Network_28054726377822
// MI455X (gfx1250) — matrix operand order repaired and checked by reading, not compiled, not run on hardware
//
#include <hip/hip_runtime.h>
#include <cmath>

typedef __attribute__((ext_vector_type(16))) __bf16 v16bf;
typedef __attribute__((ext_vector_type(8)))  __bf16 v8bf;
typedef __attribute__((ext_vector_type(4)))  __bf16 v4bf;
typedef __attribute__((ext_vector_type(8)))  float  v8f;
typedef __attribute__((ext_vector_type(4)))  float  v4f;

#define B_   64
#define T_   1024
#define D_   512
#define U_   512
#define G4U  2048
#define OUT_ 4

// ---------------------------------------------------------------------------
// Prep: pack BOTH weight matrices into exact WMMA B-fragment order (bf16),
// zero h ping-pong buffers + per-group barrier counters.
// Fragment pack layout: [n16][kt][lane(32)][elem(16)] where
//   element i of lane l:  K   = kt*32 + (l>>4)*8 + (i&7) + (i>>3)*16
//                         col = n16*16 + (l&15)
// ---------------------------------------------------------------------------
__global__ __launch_bounds__(256) void prep_kernel(
    const float* __restrict__ Wk, const float* __restrict__ Rk,
    __bf16* __restrict__ Wp, __bf16* __restrict__ Rp,
    __bf16* __restrict__ hbuf, unsigned* __restrict__ cnt)
{
  int idx = blockIdx.x * 256 + threadIdx.x;              // 0 .. 1048575
  {
    int i    = idx & 15;
    int lane = (idx >> 4) & 31;
    int kt   = (idx >> 9) & 15;
    int n16  = idx >> 13;                                // 0..127
    int krow = kt * 32 + (lane >> 4) * 8 + (i & 7) + (i >> 3) * 16;
    int col  = n16 * 16 + (lane & 15);
    size_t src = (size_t)krow * G4U + col;
    Wp[idx] = (__bf16)Wk[src];
    Rp[idx] = (__bf16)Rk[src];
  }
  if (idx < 2 * 4 * 16 * U_) hbuf[idx] = (__bf16)0.0f;   // h0 = 0
  if (idx < 4) cnt[idx] = 0u;
}

// ---------------------------------------------------------------------------
// Phase 1: XZ = X2[65536,512] @ W[512,2048] + bias  (bf16 WMMA, f32 acc)
// M rows ordered as m = t*64 + b  so each 16-row tile is one (t, batch-group).
// Output stored in C/D fragment order: XZf[(m16*128 + n16)*256 + lane*8 + e].
// Double-buffered A tile in LDS (staging overlaps WMMA, 1 barrier/k-step);
// all 4 B fragments preloaded per k-step so their loads clause together.
// ---------------------------------------------------------------------------
__global__ __launch_bounds__(256) void gemm1_kernel(
    const float* __restrict__ X, const __bf16* __restrict__ Wp,
    const float* __restrict__ bias, float* __restrict__ XZf)
{
  __shared__ __bf16 As[2][128][32];
  const int tid  = threadIdx.x;
  const int lane = tid & 31;
  const int wave = tid >> 5;
  const int wm   = wave & 3;              // M wave 0..3
  const int wn   = wave >> 2;             // N wave 0..1
  const int mblk = blockIdx.x >> 4;
  const int nblk = blockIdx.x & 15;
  const int row0 = mblk * 128;            // global m = t*64 + b
  const int l15  = lane & 15;
  const int kb   = (lane >> 4) * 8;
  const v16bf* WpV = (const v16bf*)Wp;

  // per-thread staging coordinates (4 float4 slots)
  int srow[4], scol[4];
  size_t sidx[4];
  for (int e = 0; e < 4; ++e) {
    int i4 = tid + e * 256;               // 0..1023 float4 slots
    srow[e] = i4 >> 3;
    scol[e] = (i4 & 7) * 4;
    int m = row0 + srow[e];
    int t = m >> 6, b = m & 63;
    sidx[e] = ((size_t)b * T_ + t) * D_;
  }
  auto stage = [&](int kt, int buf) {
    const int k0 = kt * 32;
    for (int e = 0; e < 4; ++e) {
      v4f f = *(const v4f*)&X[sidx[e] + k0 + scol[e]];
      *(v4bf*)&As[buf][srow[e]][scol[e]] = __builtin_convertvector(f, v4bf);
    }
  };

  v8f acc[2][4];
  for (int nt = 0; nt < 4; ++nt) {
    int n16 = nblk * 8 + wn * 4 + nt;
    float bv = bias[n16 * 16 + l15];
    for (int e = 0; e < 8; ++e) { acc[0][nt][e] = bv; acc[1][nt][e] = bv; }
  }

  stage(0, 0);
  __syncthreads();

  for (int kt = 0; kt < 16; ++kt) {
    const int cur = kt & 1;
    if (kt < 15) stage(kt + 1, cur ^ 1);  // overlaps WMMAs below

    v16bf bfrag[4];                       // 8 global_load_b128 in one clause
    for (int nt = 0; nt < 4; ++nt) {
      int n16 = nblk * 8 + wn * 4 + nt;
      bfrag[nt] = WpV[((size_t)n16 * 16 + kt) * 32 + lane];
    }
    v16bf afrag[2];
    for (int mt = 0; mt < 2; ++mt) {
      int r = wm * 32 + mt * 16 + l15;
      v8bf lo = *(const v8bf*)&As[cur][r][kb];
      v8bf hi = *(const v8bf*)&As[cur][r][kb + 16];
      for (int i = 0; i < 8; ++i) { afrag[mt][i] = lo[i]; afrag[mt][8 + i] = hi[i]; }
    }
    for (int nt = 0; nt < 4; ++nt) {
      acc[0][nt] = __builtin_amdgcn_wmma_f32_16x16x32_bf16(
          false, afrag[0], false, bfrag[nt], (short)0, acc[0][nt], false, false);
      acc[1][nt] = __builtin_amdgcn_wmma_f32_16x16x32_bf16(
          false, afrag[1], false, bfrag[nt], (short)0, acc[1][nt], false, false);
    }
    __syncthreads();                      // one barrier per k-step
  }

  for (int mt = 0; mt < 2; ++mt) {
    int m16 = mblk * 8 + wm * 2 + mt;
    for (int nt = 0; nt < 4; ++nt) {
      int n16 = nblk * 8 + wn * 4 + nt;
      *(v8f*)&XZf[((size_t)m16 * 128 + n16) * 256 + lane * 8] = acc[mt][nt];
    }
  }
}

// ---------------------------------------------------------------------------
// Phase 2: persistent LSTM scan. 64 blocks x 1 wave.
//   block = (batch group g of 16 rows) x (unit slice s of 32 hidden units)
// Per step: acc[8] (16x128 gate tile) = XZ + h @ R_slice via 128 WMMAs.
// XZ fragments: one contiguous 32B load each. R fragments: packed Rp (L2).
// c stays in VGPRs all 1024 steps; h ping-pongs through bf16 buffers in L2;
// per-group device-scope barrier via monotonic atomic counter.
// ---------------------------------------------------------------------------
__global__ __launch_bounds__(32) void lstm_kernel(
    const float* __restrict__ XZf, const __bf16* __restrict__ Rp,
    __bf16* __restrict__ hbuf, unsigned* __restrict__ cnt)
{
  const int lane = threadIdx.x;
  const int g = blockIdx.x >> 4;          // batch group 0..3
  const int s = blockIdx.x & 15;          // unit slice 0..15
  const int u0 = s * 32;
  const int l15 = lane & 15;
  const int lhi = lane >> 4;
  const int kb  = lhi * 8;

  v8f cst[2];
  for (int e = 0; e < 8; ++e) { cst[0][e] = 0.f; cst[1][e] = 0.f; }

  unsigned* mycnt = cnt + g;
  const v16bf* RpV = (const v16bf*)Rp;

  int n16tab[8];
  for (int nt = 0; nt < 8; ++nt)
    n16tab[nt] = (nt >> 1) * 32 + s * 2 + (nt & 1);     // col/16 for this slice

  for (int t = 0; t < T_; ++t) {
    const __bf16* hread  = hbuf + ((size_t)((t)     & 1) * 4 + g) * 16 * U_;
    __bf16*       hwrite = hbuf + ((size_t)((t + 1) & 1) * 4 + g) * 16 * U_;

    const float* xzb = XZf + ((size_t)(t * 4 + g) * 128) * 256 + lane * 8;
    v8f acc[8];
    for (int nt = 0; nt < 8; ++nt)
      acc[nt] = *(const v8f*)(xzb + (size_t)n16tab[nt] * 256);

    const __bf16* hp = hread + (size_t)l15 * U_;
    for (int kt = 0; kt < 16; ++kt) {
      v16bf a;
      {
        const __bf16* p = hp + kt * 32 + kb;
        v8bf lo = *(const v8bf*)p;
        v8bf hi = *(const v8bf*)(p + 16);
        for (int i = 0; i < 8; ++i) { a[i] = lo[i]; a[8 + i] = hi[i]; }
      }
      for (int nt = 0; nt < 8; ++nt) {
        v16bf b = RpV[((size_t)n16tab[nt] * 16 + kt) * 32 + lane];
        acc[nt] = __builtin_amdgcn_wmma_f32_16x16x32_bf16(
            false, a, false, b, (short)0, acc[nt], false, false);
      }
    }

    for (int p = 0; p < 2; ++p) {         // gate math + c/h update
      int col = u0 + p * 16 + l15;
      for (int e = 0; e < 8; ++e) {
        float v1 = tanhf(acc[0 + p][e]);
        float v2 = 1.f / (1.f + __expf(-acc[2 + p][e]));
        float v3 = 1.f / (1.f + __expf(-acc[4 + p][e]));
        float v4 = 1.f / (1.f + __expf(-acc[6 + p][e]));
        float cn = v1 * v2 + v3 * cst[p][e];
        cst[p][e] = cn;
        float hv = v4 * tanhf(cn);
        int m = e + lhi * 8;
        hwrite[(size_t)m * U_ + col] = (__bf16)hv;
      }
    }

    if (t + 1 < T_) {                     // prefetch next step's XZ frags
      const float* nxt = XZf + ((size_t)((t + 1) * 4 + g) * 128) * 256 + lane * 8;
      for (int nt = 0; nt < 8; ++nt)
        __builtin_prefetch(nxt + (size_t)n16tab[nt] * 256, 0, 0);
    }

    __threadfence();                      // make h visible device-wide
    if (lane == 0)
      __hip_atomic_fetch_add(mycnt, 1u, __ATOMIC_RELEASE, __HIP_MEMORY_SCOPE_AGENT);
    const unsigned target = 16u * (unsigned)(t + 1);
    while (__hip_atomic_load(mycnt, __ATOMIC_ACQUIRE, __HIP_MEMORY_SCOPE_AGENT) < target)
      __builtin_amdgcn_s_sleep(1);
  }
}

// ---------------------------------------------------------------------------
// Head: out = softmax(h_last @ fc_w + fc_b). h_last lives in parity-0 buffer
// (written at t=1023 into buffer (1024 & 1) == 0).
// ---------------------------------------------------------------------------
__global__ __launch_bounds__(64) void head_kernel(
    const __bf16* __restrict__ hlast, const float* __restrict__ fcw,
    const float* __restrict__ fcb, float* __restrict__ out)
{
  int b = threadIdx.x;
  const __bf16* hp = hlast + (size_t)b * U_;   // (g*16+row)*512 is contiguous
  float a0 = fcb[0], a1 = fcb[1], a2 = fcb[2], a3 = fcb[3];
  for (int k = 0; k < U_; ++k) {
    float hv = (float)hp[k];
    a0 += hv * fcw[k * 4 + 0]; a1 += hv * fcw[k * 4 + 1];
    a2 += hv * fcw[k * 4 + 2]; a3 += hv * fcw[k * 4 + 3];
  }
  float mx = fmaxf(fmaxf(a0, a1), fmaxf(a2, a3));
  float e0 = __expf(a0 - mx), e1 = __expf(a1 - mx);
  float e2 = __expf(a2 - mx), e3 = __expf(a3 - mx);
  float sum = e0 + e1 + e2 + e3;
  out[b * 4 + 0] = e0 / sum; out[b * 4 + 1] = e1 / sum;
  out[b * 4 + 2] = e2 / sum; out[b * 4 + 3] = e3 / sum;
}

extern "C" void kernel_launch(void* const* d_in, const int* in_sizes, int n_in,
                              void* d_out, int out_size, void* d_ws, size_t ws_size,
                              hipStream_t stream) {
  const float* tx   = (const float*)d_in[0];
  const float* Wk   = (const float*)d_in[1];
  const float* Rk   = (const float*)d_in[2];
  const float* bias = (const float*)d_in[3];
  const float* fcw  = (const float*)d_in[4];
  const float* fcb  = (const float*)d_in[5];
  float* out = (float*)d_out;

  char* ws = (char*)d_ws;
  size_t off = 0;
  auto take = [&](size_t bytes) -> char* {
    char* p = ws + off;
    off = (off + bytes + 255) & ~(size_t)255;
    return p;
  };
  float*    XZf = (float*)   take((size_t)B_ * T_ * G4U * sizeof(float)); // 512 MB
  __bf16*   Wp  = (__bf16*)  take((size_t)D_ * G4U * 2);                  // 2 MB
  __bf16*   Rp  = (__bf16*)  take((size_t)U_ * G4U * 2);                  // 2 MB
  __bf16*   hb  = (__bf16*)  take((size_t)2 * 4 * 16 * U_ * 2);           // 128 KB
  unsigned* cnt = (unsigned*)take(256);

  prep_kernel <<<4096, 256, 0, stream>>>(Wk, Rk, Wp, Rp, hb, cnt);
  gemm1_kernel<<<8192, 256, 0, stream>>>(tx, Wp, bias, XZf);
  lstm_kernel <<<64,   32,  0, stream>>>(XZf, Rp, hb, cnt);
  head_kernel <<<1,    64,  0, stream>>>(hb, fcw, fcb, out);
}
